// Module_79869211836507
// MI455X (gfx1250) — matrix operand order repaired and checked by reading, not compiled, not run on hardware
//
#include <hip/hip_runtime.h>
#include <hip/hip_bf16.h>

// ---------------------------------------------------------------------------
// Problem constants (match reference)
// ---------------------------------------------------------------------------
#define UU    20000      // users
#define II    10000      // items
#define LDA   (II + 1)   // interactions row stride (floats)
#define BB    4096       // batch
#define KDIM  64         // embedding dim K
#define IPAD  10016      // II padded to multiple of 32 (zero-padded weights)

#define UCHUNKS 4        // split-K for ugemm (K = IPAD)
#define ICHUNKS 8        // split-K for igemm (K = UU)
#define UCHUNK  2560     // 3x2560 + 2336, all multiples of 32
#define ICHUNK  2496     // 7x2496 + 2528, all multiples of 32

typedef __attribute__((ext_vector_type(16))) __bf16        v16bf;
typedef __attribute__((ext_vector_type(8)))  float         v8f;
typedef __attribute__((ext_vector_type(8)))  unsigned short u16x8;

union BF16Frag {
    v16bf bf;
    unsigned short u[16];
    unsigned int   p[8];
};

// RNE conversion: used only in the one-shot weight conversion kernel.
__device__ __forceinline__ unsigned short f2bf(float f) {
    unsigned int u = __builtin_bit_cast(unsigned int, f);
    unsigned int r = u + 0x7FFFu + ((u >> 16) & 1u);
    return (unsigned short)(r >> 16);
}

// Pack two f32 into one bf16x2 dword by truncation (exact for 0.0/1.0
// interaction values): one v_perm_b32.
__device__ __forceinline__ unsigned int pack_bf16(float lo, float hi) {
    return __builtin_amdgcn_perm(__builtin_bit_cast(unsigned int, hi),
                                 __builtin_bit_cast(unsigned int, lo),
                                 0x07060302u);
}

// ---------------------------------------------------------------------------
// Kernel 1: convert weights to bf16 (W_u zero-padded in K to IPAD)
//   Wub: [64, IPAD]  bf16 row-major,  Wib: [64, UU] bf16 row-major
// ---------------------------------------------------------------------------
__global__ void convw(const float* __restrict__ Wu, const float* __restrict__ Wi,
                      unsigned short* __restrict__ Wub, unsigned short* __restrict__ Wib) {
    const long nu = 64L * IPAD;     // 641024
    const long ni = 64L * UU;       // 1280000
    long i = (long)blockIdx.x * blockDim.x + threadIdx.x;
    if (i < nu) {
        int n = (int)(i / IPAD), k = (int)(i % IPAD);
        Wub[i] = (k < II) ? f2bf(Wu[(long)n * II + k]) : (unsigned short)0;
    } else if (i < nu + ni) {
        long j = i - nu;
        Wib[j] = f2bf(Wi[j]);
    }
}

// ---------------------------------------------------------------------------
// Kernel 2: partial u_emb = gather(rows)[4096, kk0:kkn] @ W_u^T chunk
// One wave per (16-row M tile x K chunk); 4 accumulators cover N = 64.
// grid = (BB/16, UCHUNKS)
// ---------------------------------------------------------------------------
__global__ __launch_bounds__(32) void ugemm(const float* __restrict__ inter,
                                            const int* __restrict__ uidx,
                                            const unsigned short* __restrict__ Wub,
                                            float* __restrict__ partU) {
    const int lane = threadIdx.x;          // 0..31
    const int m    = lane & 15;
    const int h    = lane >> 4;
    const int mt   = blockIdx.x;           // 0..255 (4096/16)
    const int ch   = blockIdx.y;           // K chunk
    const int kk0  = ch * UCHUNK;
    const int kkn  = (kk0 + UCHUNK < IPAD) ? kk0 + UCHUNK : IPAD;
    float* __restrict__ outp = partU + (long)ch * BB * KDIM;

    const long row = (long)uidx[mt * 16 + m];
    const float* __restrict__ aRow = inter + row * (long)LDA;

    v8f acc[4] = {v8f{}, v8f{}, v8f{}, v8f{}};

    for (int kk = kk0; kk < kkn; kk += 32) {
        const int k0 = kk + h * 8;
        __builtin_prefetch(aRow + k0 + 512, 0, 1);   // global_prefetch_b8

        // A fragment: lane half h holds K = 8h+{0..7} and 16+8h+{0..7};
        // pack f32 pairs -> bf16x2 dwords via v_perm (truncation exact, A is 0/1)
        BF16Frag a;
        #pragma unroll
        for (int j = 0; j < 8; j += 2) {
            a.p[j >> 1]       = pack_bf16(aRow[k0 + j],      aRow[k0 + j + 1]);
            a.p[(j >> 1) + 4] = pack_bf16(aRow[k0 + 16 + j], aRow[k0 + 17 + j]);
        }

        const int kb = kk + h * 8;
        BF16Frag b[4];
        #pragma unroll
        for (int nt = 0; nt < 4; ++nt) {
            const u16x8* __restrict__ wp =
                (const u16x8*)(Wub + (long)(nt * 16 + m) * IPAD + kb);
            *(u16x8*)&b[nt].u[0] = wp[0];
            *(u16x8*)&b[nt].u[8] = wp[2];
        }
        #pragma unroll
        for (int nt = 0; nt < 4; ++nt) {
            acc[nt] = __builtin_amdgcn_wmma_f32_16x16x32_bf16(
                false, a.bf, false, b[nt].bf, (short)0, acc[nt], false, false);
        }
    }

    // C layout: VGPR r -> M = r + 8h, N = lane%16
    #pragma unroll
    for (int nt = 0; nt < 4; ++nt) {
        #pragma unroll
        for (int r = 0; r < 8; ++r) {
            const int mr = mt * 16 + r + h * 8;
            outp[(long)mr * KDIM + nt * 16 + m] = acc[nt][r];
        }
    }
}

// ---------------------------------------------------------------------------
// Kernel 3: partial e_i = inter[kk0:kkn, 0:I]^T @ W_i^T chunk -> [IPAD, 64]
// Streams the 800 MB interactions matrix exactly once (across chunks).
// Double-buffered async global->LDS copies (ASYNCcnt), raw f32 tiles in LDS
// (row stride 20 floats = 80B: 16B-aligned async dests, conflict-free reads).
// grid = (IPAD/16, ICHUNKS)
// ---------------------------------------------------------------------------
#define TSTRIDE 20   // floats per LDS tile row (80 bytes)

__device__ __forceinline__ void issue_tile_async(const float* __restrict__ inter,
                                                 long jbase, int kk,
                                                 float* lbuf, int lane) {
    // lane copies its row (kk+lane), 16 consecutive floats, as 4x async b128.
    const float* g = inter + (long)(kk + lane) * LDA + jbase;
    unsigned l = (unsigned)(unsigned long long)(lbuf + lane * TSTRIDE); // LDS byte offset = addr[31:0]
    asm volatile(
        "s_wait_dscnt 0x0\n\t"
        "global_load_async_to_lds_b128 %0, %2, off\n\t"
        "global_load_async_to_lds_b128 %1, %3, off"
        :
        : "v"(l), "v"(l + 32), "v"(g), "v"(g + 8)
        : "memory");
    asm volatile(
        "global_load_async_to_lds_b128 %0, %2, off\n\t"
        "global_load_async_to_lds_b128 %1, %3, off"
        :
        : "v"(l + 16), "v"(l + 48), "v"(g + 4), "v"(g + 12)
        : "memory");
}

__global__ __launch_bounds__(32) void igemm(const float* __restrict__ inter,
                                            const unsigned short* __restrict__ Wib,
                                            float* __restrict__ partI) {
    __shared__ __align__(16) float atile[2][32 * TSTRIDE];   // 2 x 2560 B

    const int lane = threadIdx.x;
    const int m    = lane & 15;
    const int h    = lane >> 4;
    const long jbase = (long)blockIdx.x * 16;      // item tile base (0..10000)
    const int ch   = blockIdx.y;                   // K chunk
    const int kk0  = ch * ICHUNK;
    const int kkn  = (ch == ICHUNKS - 1) ? UU : kk0 + ICHUNK;
    float* __restrict__ outp = partI + (long)ch * IPAD * KDIM;

    v8f acc[4] = {v8f{}, v8f{}, v8f{}, v8f{}};

    issue_tile_async(inter, jbase, kk0, atile[0], lane);

    for (int kk = kk0; kk < kkn; kk += 32) {
        const int bi = ((kk - kk0) >> 5) & 1;
        if (kk + 32 < kkn) {
            issue_tile_async(inter, jbase, kk + 32, atile[bi ^ 1], lane);
            // async loads complete in order: <=4 outstanding => current tile landed
            asm volatile("s_wait_asynccnt 0x4" ::: "memory");
        } else {
            asm volatile("s_wait_asynccnt 0x0" ::: "memory");
        }

        // A fragment from LDS f32 tile (transposed access): column j = m,
        // K (= u) = 8h+{0..7} and 16+8h+{0..7}; pack pairs via v_perm.
        const float* tb = atile[bi];
        BF16Frag a;
        #pragma unroll
        for (int j = 0; j < 8; j += 2) {
            const int u0 = h * 8 + j;
            a.p[j >> 1]       = pack_bf16(tb[(u0)      * TSTRIDE + m],
                                          tb[(u0 + 1)  * TSTRIDE + m]);
            a.p[(j >> 1) + 4] = pack_bf16(tb[(u0 + 16) * TSTRIDE + m],
                                          tb[(u0 + 17) * TSTRIDE + m]);
        }

        const int kb = kk + h * 8;
        BF16Frag b[4];
        #pragma unroll
        for (int nt = 0; nt < 4; ++nt) {
            const u16x8* __restrict__ wp =
                (const u16x8*)(Wib + (long)(nt * 16 + m) * UU + kb);
            *(u16x8*)&b[nt].u[0] = wp[0];
            *(u16x8*)&b[nt].u[8] = wp[2];
        }
        #pragma unroll
        for (int nt = 0; nt < 4; ++nt) {
            acc[nt] = __builtin_amdgcn_wmma_f32_16x16x32_bf16(
                false, a.bf, false, b[nt].bf, (short)0, acc[nt], false, false);
        }
    }

    #pragma unroll
    for (int nt = 0; nt < 4; ++nt) {
        #pragma unroll
        for (int r = 0; r < 8; ++r) {
            const long jr = jbase + r + h * 8;
            outp[jr * KDIM + nt * 16 + m] = acc[nt][r];
        }
    }
}

// ---------------------------------------------------------------------------
// Kernel 3b: deterministic fixed-order reduction of split-K partials.
// ---------------------------------------------------------------------------
__global__ void reduceparts(const float* __restrict__ partU,
                            const float* __restrict__ partI,
                            float* __restrict__ uemb, float* __restrict__ ei) {
    const long NUE = (long)BB * KDIM;     // 262144
    const long NIE = (long)IPAD * KDIM;   // 641024
    long i = (long)blockIdx.x * blockDim.x + threadIdx.x;
    if (i < NUE) {
        float s = 0.f;
        #pragma unroll
        for (int c = 0; c < UCHUNKS; ++c) s += partU[(long)c * NUE + i];
        uemb[i] = s;
    } else if (i < NUE + NIE) {
        const long j = i - NUE;
        float s = 0.f;
        #pragma unroll
        for (int c = 0; c < ICHUNKS; ++c) s += partI[(long)c * NIE + j];
        ei[j] = s;
    }
}

// ---------------------------------------------------------------------------
// Kernel 4: epilogue — mask correction, concat, 3x (Linear+LN+ReLU), project.
// One wave per batch element; per-wave LDS scratch, sync-free (wave-private DS).
// ---------------------------------------------------------------------------
__global__ __launch_bounds__(256) void mlp(
    const float* __restrict__ inter,
    const int* __restrict__ uidx, const int* __restrict__ iidx,
    const float* __restrict__ Wu, const float* __restrict__ Wi,
    const float* __restrict__ uemb, const float* __restrict__ ei,
    const float* __restrict__ W1, const float* __restrict__ b1,
    const float* __restrict__ g1, const float* __restrict__ be1,
    const float* __restrict__ W2, const float* __restrict__ b2,
    const float* __restrict__ g2, const float* __restrict__ be2,
    const float* __restrict__ W3, const float* __restrict__ b3,
    const float* __restrict__ g3, const float* __restrict__ be3,
    const float* __restrict__ Wp, const float* __restrict__ bp,
    float* __restrict__ out) {

    __shared__ float xs[8][128];
    __shared__ float hs[8][64];

    const int lane = threadIdx.x & 31;
    const int w    = threadIdx.x >> 5;
    const int b    = blockIdx.x * 8 + w;

    const int us = uidx[b];
    const int it = iidx[b];
    const float mval = inter[(long)us * LDA + it];   // masked element (0 or 1)

    float* x    = xs[w];
    float* hbuf = hs[w];

    // concat [u_emb - m*W_u[:,it],  e_i[it] - m*W_i[:,us]]
    #pragma unroll
    for (int r = 0; r < 4; ++r) {
        const int idx = lane + 32 * r;
        float v;
        if (idx < 64) {
            v = uemb[(long)b * KDIM + idx] - mval * Wu[(long)idx * II + it];
        } else {
            const int k = idx - 64;
            v = ei[(long)it * KDIM + k] - mval * Wi[(long)k * UU + us];
        }
        x[idx] = v;
    }

    const float* Ws[3]  = {W1, W2, W3};
    const float* bs[3]  = {b1, b2, b3};
    const float* gs[3]  = {g1, g2, g3};
    const float* bes[3] = {be1, be2, be3};
    const int din[3]  = {128, 64, 32};
    const int dout[3] = {64, 32, 16};

    for (int l = 0; l < 3; ++l) {
        const int in = din[l], on = dout[l];
        const float* __restrict__ W = Ws[l];
        for (int o = lane; o < on; o += 32) {
            float acc = bs[l][o];
            for (int i = 0; i < in; ++i) acc += x[i] * W[o * in + i];
            hbuf[o] = acc;
        }
        // LayerNorm stats (redundant per lane; tiny)
        float s = 0.f, s2 = 0.f;
        for (int i = 0; i < on; ++i) { float v = hbuf[i]; s += v; s2 += v * v; }
        const float mean = s / on;
        const float var  = s2 / on - mean * mean;
        const float rstd = rsqrtf(var + 1e-5f);
        for (int o = lane; o < on; o += 32) {
            const float y = (hbuf[o] - mean) * rstd * gs[l][o] + bes[l][o];
            x[o] = fmaxf(y, 0.f);
        }
    }

    if (lane == 0) {
        float acc = bp[0];
        for (int k = 0; k < 16; ++k) acc += x[k] * Wp[k];
        out[b] = acc;
    }
}

// ---------------------------------------------------------------------------
// Launch
// ---------------------------------------------------------------------------
extern "C" void kernel_launch(void* const* d_in, const int* in_sizes, int n_in,
                              void* d_out, int out_size, void* d_ws, size_t ws_size,
                              hipStream_t stream) {
    const float* inter = (const float*)d_in[0];
    const int*   uidx  = (const int*)d_in[1];
    const int*   iidx  = (const int*)d_in[2];
    const float* Wu    = (const float*)d_in[3];
    const float* Wi    = (const float*)d_in[4];
    const float* W1  = (const float*)d_in[5];
    const float* b1  = (const float*)d_in[6];
    const float* g1  = (const float*)d_in[7];
    const float* be1 = (const float*)d_in[8];
    const float* W2  = (const float*)d_in[9];
    const float* b2  = (const float*)d_in[10];
    const float* g2  = (const float*)d_in[11];
    const float* be2 = (const float*)d_in[12];
    const float* W3  = (const float*)d_in[13];
    const float* b3  = (const float*)d_in[14];
    const float* g3  = (const float*)d_in[15];
    const float* be3 = (const float*)d_in[16];
    const float* Wp  = (const float*)d_in[17];
    const float* bp  = (const float*)d_in[18];

    // workspace layout (all offsets 256B aligned), total ~31 MB
    char* ws = (char*)d_ws;
    unsigned short* Wub   = (unsigned short*)(ws);               // 1,282,048 B
    unsigned short* Wib   = (unsigned short*)(ws + 1282048);     // 2,560,000 B
    float*          uemb  = (float*)(ws + 3842048);              // 1,048,576 B
    float*          ei    = (float*)(ws + 4890624);              // 2,564,096 B
    float*          partU = (float*)(ws + 7454720);              // 4 x 1,048,576 = 4,194,304 B
    float*          partI = (float*)(ws + 11649024);             // 8 x 2,564,096 = 20,512,768 B

    // 1) weights -> bf16 (padded)
    {
        const long total = 64L * IPAD + 64L * UU;   // 1,921,024
        const int blocks = (int)((total + 255) / 256);
        convw<<<blocks, 256, 0, stream>>>(Wu, Wi, Wub, Wib);
    }
    // 2) user-side GEMM, split-K over 4 chunks: 1024 waves
    ugemm<<<dim3(BB / 16, UCHUNKS), 32, 0, stream>>>(inter, uidx, Wub, partU);
    // 3) item-side GEMM, split-K over 8 chunks: 5008 waves (streams matrix once)
    igemm<<<dim3(IPAD / 16, ICHUNKS), 32, 0, stream>>>(inter, Wib, partI);
    // 3b) deterministic split-K reduction
    {
        const long total = (long)BB * KDIM + (long)IPAD * KDIM;  // 903,168
        const int blocks = (int)((total + 255) / 256);
        reduceparts<<<blocks, 256, 0, stream>>>(partU, partI, uemb, ei);
    }
    // 4) epilogue: mask fixup + MLP
    mlp<<<BB / 8, 256, 0, stream>>>(inter, uidx, iidx, Wu, Wi, uemb, ei,
                                    W1, b1, g1, be1, W2, b2, g2, be2,
                                    W3, b3, g3, be3, Wp, bp, (float*)d_out);
}
